// Generator_51350628991081
// MI455X (gfx1250) — matrix operand order repaired and checked by reading, not compiled, not run on hardware
//
#include <hip/hip_runtime.h>
#include <hip/hip_bf16.h>
#include <math.h>

// ---------------------------------------------------------------------------
// Problem constants (fixed by the reference setup)
// ---------------------------------------------------------------------------
#define CB   32          // batch
#define CS   400         // encoder seq len
#define CH   512         // hidden
#define CH2  1024        // 2H
#define CH3  1536        // 3H
#define CE   300         // embed dim
#define CVS  200         // user-specific vocab
#define CT   32          // target steps
#define CEN  50000       // embed_num (gen vocab)
#define CV   50200       // padded vocab
#define CKC  2136        // 3H + 2E  (concat for Wc)
#define CKG  1836        // 2H + H + E (concat for gate)
#define NT_GEN 3125      // CEN / 16 n-tiles for the Wgen GEMM
#define NT_PAD 3128      // padded to multiple of NTW so OOB B-loads stay in-bounds
#define KB_GEN 16        // CH / 32 k-blocks
#define NTW  4           // n-tiles per wave in the hot GEMM (16x64 per wave)

typedef __attribute__((ext_vector_type(16))) __bf16 v16bf;
typedef __attribute__((ext_vector_type(8)))  float  v8f;

union FragAB { v16bf v; __bf16 e[16]; };
union FragC  { v8f   v; float  e[8];  };

// 16-bit A-matrix 16x32 per-lane K index (ISA 7.12.2): VGPR i, lane-half h
__device__ __forceinline__ int kA16(int i, int half) {
    return (i < 4) ? (2 * i + 8 * half) : (16 + 2 * (i - 4) + 8 * half);
}
// 16-bit B-matrix 32x16: VGPR i, lane-half h -> K base (same order as A)
__device__ __forceinline__ int kB16(int i, int half) {
    return (i < 4) ? (2 * i + 8 * half) : (16 + 2 * (i - 4) + 8 * half);
}

__device__ __forceinline__ float sigmf(float x) { return 1.0f / (1.0f + expf(-x)); }

// Branchless guarded load: clamp the index (always-safe address) then select.
__device__ __forceinline__ float ldg_guard(const float* __restrict__ p, size_t rowBase,
                                           int k, int stride, int K) {
    int kc = (k < K) ? k : (K - 1);                 // v_cndmask on index
    float v = p[rowBase + (size_t)kc * stride];     // unconditional load
    return (k < K) ? v : 0.0f;                      // v_cndmask on value
}

// ---------------------------------------------------------------------------
// Generic bf16 WMMA GEMM:  C[M,N] = act( A[M,K] @ B + bias )
//   TRANSB==0 : B stored [K,N] row-major
//   TRANSB==1 : B stored [N,K] row-major (i.e. compute A @ B^T)
// Compile-time TRANSB/ACT -> straight-line inner loop (no uniform branches).
// A,B are f32 in memory, converted to bf16 while building fragments.
// One wave per 16x16 output tile; branchless zero-guards handle odd K.
// ---------------------------------------------------------------------------
template <int TRANSB, int ACT>
__global__ void wmma_gemm(const float* __restrict__ A, const float* __restrict__ Bm,
                          const float* __restrict__ bias, float* __restrict__ C,
                          int M, int N, int K) {
    const int lane = threadIdx.x & 31;
    const int wi   = blockIdx.x * (blockDim.x >> 5) + (threadIdx.x >> 5);
    const int nTiles = N >> 4;
    const int tiles  = (M >> 4) * nTiles;
    if (wi >= tiles) return;
    const int mt = wi / nTiles;
    const int nt = wi % nTiles;
    const int m0 = mt << 4, n0 = nt << 4;
    const int m  = lane & 15, half = lane >> 4;

    const size_t aRow  = (size_t)(m0 + m) * K;      // A row base for this lane
    const size_t bRowT = (size_t)(n0 + m) * K;      // B^T row base (TRANSB)

    FragC acc;
    #pragma unroll
    for (int i = 0; i < 8; ++i) acc.e[i] = 0.0f;

    for (int kk = 0; kk < K; kk += 32) {
        FragAB a, b;
        #pragma unroll
        for (int i = 0; i < 8; ++i) {
            int ka = kk + kA16(i, half);
            a.e[2*i]   = (__bf16)ldg_guard(A, aRow, ka,     1, K);
            a.e[2*i+1] = (__bf16)ldg_guard(A, aRow, ka + 1, 1, K);
            int kb = kk + kB16(i, half);
            float b0, b1;
            if (TRANSB) {
                b0 = ldg_guard(Bm, bRowT, kb,     1, K);
                b1 = ldg_guard(Bm, bRowT, kb + 1, 1, K);
            } else {
                b0 = ldg_guard(Bm, (size_t)(n0 + m), kb,     N, K);
                b1 = ldg_guard(Bm, (size_t)(n0 + m), kb + 1, N, K);
            }
            b.e[2*i]   = (__bf16)b0;
            b.e[2*i+1] = (__bf16)b1;
        }
        acc.v = __builtin_amdgcn_wmma_f32_16x16x32_bf16(
            false, a.v, false, b.v, (short)0, acc.v, false, false);
    }

    #pragma unroll
    for (int i = 0; i < 8; ++i) {
        int row = m0 + i + 8 * half;
        int col = n0 + (lane & 15);
        float v = acc.e[i];
        if (bias) v += bias[col];
        if (ACT == 1) v = tanhf(v);
        C[(size_t)row * N + col] = v;
    }
}

// ---------------------------------------------------------------------------
// Pack Wgen [512, 50000] f32 -> bf16 WMMA B-fragments, layout [kb][nt][lane][16]
// (nt stride padded to NT_PAD) so the hot GEMM loads one contiguous 32B vector
// per lane per k-step.
// ---------------------------------------------------------------------------
__global__ void pack_wgen(const float* __restrict__ Wgen, __bf16* __restrict__ dst) {
    const long long total = (long long)KB_GEN * NT_GEN * 32 * 16;
    long long idx = (long long)blockIdx.x * blockDim.x + threadIdx.x;
    if (idx >= total) return;
    int j    = (int)(idx & 15);
    int lane = (int)((idx >> 4) & 31);
    int nt   = (int)((idx >> 9) % NT_GEN);
    int kb   = (int)(idx / ((long long)NT_GEN * 512));
    int n    = lane & 15, half = lane >> 4;
    int i    = j >> 1;
    int k    = kb * 32 + kB16(i, half) + (j & 1);
    int col  = nt * 16 + n;
    size_t off = (((size_t)kb * NT_PAD + nt) * 32 + lane) * 16 + j;
    dst[off] = (__bf16)Wgen[(size_t)k * CEN + col];
}

// ---------------------------------------------------------------------------
// Hot per-step GEMM: logits[32,50000] = chb[32,512] @ Wgen(bf16 packed)
// Each wave computes a 16x64 strip (NTW=4 n-tiles): one A fragment per
// K-block feeds 4 back-to-back WMMAs; B fragments are contiguous 32B vector
// loads from the L2-resident packed buffer, with prefetch of the next K-block.
// ---------------------------------------------------------------------------
__global__ void wgen_gemm(const float* __restrict__ A, const __bf16* __restrict__ Bp,
                          float* __restrict__ Cl) {
    const int lane = threadIdx.x & 31;
    const int wave = threadIdx.x >> 5;
    const int ntBase = (blockIdx.x * (blockDim.x >> 5) + wave) * NTW;
    const int mt   = blockIdx.y;
    if (ntBase >= NT_GEN) return;
    const int m0 = mt << 4;
    const int m  = lane & 15, half = lane >> 4;

    FragC acc[NTW];
    #pragma unroll
    for (int j = 0; j < NTW; ++j)
        #pragma unroll
        for (int i = 0; i < 8; ++i) acc[j].e[i] = 0.0f;

    const __bf16* bbase = Bp + ((size_t)ntBase * 32 + lane) * 16;
    const size_t  kbStride = (size_t)NT_PAD * 512;
    const float*  aRow = A + (m0 + m) * CH;

    for (int kb = 0; kb < KB_GEN; ++kb) {
        FragAB a;
        const int kk = kb * 32;
        #pragma unroll
        for (int i = 0; i < 8; ++i) {
            int ka = kk + kA16(i, half);
            a.e[2*i]   = (__bf16)aRow[ka];
            a.e[2*i+1] = (__bf16)aRow[ka + 1];
        }
        const __bf16* bkb = bbase + (size_t)kb * kbStride;
        if (kb + 1 < KB_GEN)
            __builtin_prefetch(bbase + (size_t)(kb + 1) * kbStride, 0, 1);
        #pragma unroll
        for (int j = 0; j < NTW; ++j) {
            FragAB b;
            b.v = *(const v16bf*)(bkb + (size_t)j * (32 * 16));
            acc[j].v = __builtin_amdgcn_wmma_f32_16x16x32_bf16(
                false, a.v, false, b.v, (short)0, acc[j].v, false, false);
        }
    }

    #pragma unroll
    for (int j = 0; j < NTW; ++j) {
        int nt = ntBase + j;
        if (nt < NT_GEN) {                           // wave-uniform guard
            #pragma unroll
            for (int i = 0; i < 8; ++i) {
                int row = m0 + i + 8 * half;
                int col = (nt << 4) + (lane & 15);
                Cl[(size_t)row * CEN + col] = acc[j].e[i];
            }
        }
    }
}

// ---------------------------------------------------------------------------
// Small per-step kernels (VALU work — negligible vs the roofline)
// ---------------------------------------------------------------------------
__global__ void gather_prev(const float* __restrict__ Wemb, const int* __restrict__ tgt,
                            float* __restrict__ prev) {
    int idx = blockIdx.x * blockDim.x + threadIdx.x;          // T*B*E
    if (idx >= CT * CB * CE) return;
    int e  = idx % CE;
    int tb = idx / CE;
    int b  = tb % CB;
    int t  = tb / CB;
    int tok = (t == 0) ? 1 : tgt[b * CT + (t - 1)];
    prev[idx] = Wemb[(size_t)tok * CE + e];
}

__global__ void gather_uve(const float* __restrict__ Wemb, const int* __restrict__ sv,
                           float* __restrict__ uve) {
    int idx = blockIdx.x * blockDim.x + threadIdx.x;          // B*Vs*E
    if (idx >= CB * CVS * CE) return;
    int e = idx % CE;
    int r = idx / CE;
    int tok = sv[r];
    uve[idx] = Wemb[(size_t)tok * CE + e];
}

__global__ void gru_fuse(const float* __restrict__ gi, const float* __restrict__ gh,
                         float* __restrict__ h) {
    int idx = blockIdx.x * blockDim.x + threadIdx.x;          // B*H
    if (idx >= CB * CH) return;
    int b = idx / CH, j = idx % CH;
    const float* gib = gi + (size_t)b * CH3;
    const float* ghb = gh + (size_t)b * CH3;
    float r = sigmf(gib[j]            + ghb[j]);
    float z = sigmf(gib[CH + j]       + ghb[CH + j]);
    float n = tanhf(gib[2 * CH + j] + r * ghb[2 * CH + j]);
    h[idx] = (1.0f - z) * n + z * h[idx];
}

// sc[row] = tanh(qp[b,:] + pk[row,:]) . we + be   (H=512, one wave per row)
__global__ void att_score(const float* __restrict__ qp, const float* __restrict__ pk,
                          const float* __restrict__ we, const float* __restrict__ be,
                          float* __restrict__ sc, int S) {
    int lane = threadIdx.x & 31;
    int row  = blockIdx.x * (blockDim.x >> 5) + (threadIdx.x >> 5);
    if (row >= CB * S) return;
    int b = row / S;
    float sum = 0.0f;
    #pragma unroll
    for (int j = lane; j < CH; j += 32)
        sum += tanhf(qp[b * CH + j] + pk[(size_t)row * CH + j]) * we[j];
    #pragma unroll
    for (int off = 16; off > 0; off >>= 1) sum += __shfl_xor(sum, off, 32);
    if (lane == 0) sc[row] = sum + be[0];
}

__global__ void softmax_row(const float* __restrict__ sc, const int* __restrict__ mask,
                            float* __restrict__ a, int S) {
    __shared__ float red[256];
    int b = blockIdx.x, tid = threadIdx.x;
    float mx = -3.4e38f;
    for (int s = tid; s < S; s += 256) {
        float v = sc[b * S + s];
        if (mask && mask[b * S + s] == 0) v = -1e9f;
        mx = fmaxf(mx, v);
    }
    red[tid] = mx; __syncthreads();
    for (int o = 128; o > 0; o >>= 1) {
        if (tid < o) red[tid] = fmaxf(red[tid], red[tid + o]);
        __syncthreads();
    }
    mx = red[0]; __syncthreads();
    float sum = 0.0f;
    for (int s = tid; s < S; s += 256) {
        float v = sc[b * S + s];
        if (mask && mask[b * S + s] == 0) v = -1e9f;
        sum += expf(v - mx);
    }
    red[tid] = sum; __syncthreads();
    for (int o = 128; o > 0; o >>= 1) {
        if (tid < o) red[tid] += red[tid + o];
        __syncthreads();
    }
    float inv = 1.0f / red[0];
    for (int s = tid; s < S; s += 256) {
        float v = sc[b * S + s];
        if (mask && mask[b * S + s] == 0) v = -1e9f;
        a[b * S + s] = expf(v - mx) * inv;
    }
}

// ctx[b,d] = sum_s a[b,s] * X[b,s,d]
__global__ void ctx_comb(const float* __restrict__ a, const float* __restrict__ X,
                         float* __restrict__ ctx, int S, int D) {
    int b = blockIdx.y;
    int d = blockIdx.x * blockDim.x + threadIdx.x;
    if (d >= D) return;
    float acc = 0.0f;
    const float* ab = a + b * S;
    const float* Xb = X + (size_t)b * S * D + d;
    for (int s = 0; s < S; ++s) acc += ab[s] * Xb[(size_t)s * D];
    ctx[b * D + d] = acc;
}

__global__ void build_concat(const float* __restrict__ h, const float* __restrict__ ctxc,
                             const float* __restrict__ ctxu, const float* __restrict__ ue,
                             float* __restrict__ cc) {
    int idx = blockIdx.x * blockDim.x + threadIdx.x;          // B*2136
    if (idx >= CB * CKC) return;
    int b = idx / CKC, c = idx % CKC;
    float v;
    if      (c < CH)              v = h[b * CH + c];
    else if (c < CH + CH2)        v = ctxc[b * CH2 + (c - CH)];
    else if (c < CH + CH2 + CE)   v = ctxu[b * CE + (c - CH - CH2)];
    else                          v = ue[b * CE + (c - CH - CH2 - CE)];
    cc[idx] = v;
}

// cp[b] = sigmoid( [ctx_c, h, ctx_u] . Wg + bg )
__global__ void gate_prob(const float* __restrict__ ctxc, const float* __restrict__ h,
                          const float* __restrict__ ctxu, const float* __restrict__ Wg,
                          const float* __restrict__ bg, float* __restrict__ cp) {
    __shared__ float red[256];
    int b = blockIdx.x, tid = threadIdx.x;
    float sum = 0.0f;
    for (int i = tid; i < CKG; i += 256) {
        float x;
        if      (i < CH2)       x = ctxc[b * CH2 + i];
        else if (i < CH2 + CH)  x = h[b * CH + (i - CH2)];
        else                    x = ctxu[b * CE + (i - CH2 - CH)];
        sum += Wg[i] * x;
    }
    red[tid] = sum; __syncthreads();
    for (int o = 128; o > 0; o >>= 1) {
        if (tid < o) red[tid] += red[tid + o];
        __syncthreads();
    }
    if (tid == 0) cp[b] = sigmf(red[0] + bg[0]);
}

__global__ void row_stats(const float* __restrict__ logits, float* __restrict__ mx,
                          float* __restrict__ sinv) {
    __shared__ float red[256];
    int b = blockIdx.x, tid = threadIdx.x;
    const float* lb = logits + (size_t)b * CEN;
    float m = -3.4e38f;
    for (int v = tid; v < CEN; v += 256) m = fmaxf(m, lb[v]);
    red[tid] = m; __syncthreads();
    for (int o = 128; o > 0; o >>= 1) {
        if (tid < o) red[tid] = fmaxf(red[tid], red[tid + o]);
        __syncthreads();
    }
    m = red[0]; __syncthreads();
    float s = 0.0f;
    for (int v = tid; v < CEN; v += 256) s += __expf(lb[v] - m);
    red[tid] = s; __syncthreads();
    for (int o = 128; o > 0; o >>= 1) {
        if (tid < o) red[tid] += red[tid + o];
        __syncthreads();
    }
    if (tid == 0) { mx[b] = m; sinv[b] = 1.0f / red[0]; }
}

__global__ void mix_write(const float* __restrict__ logits, const float* __restrict__ mx,
                          const float* __restrict__ sinv, const float* __restrict__ cp,
                          float* __restrict__ out, int t) {
    int b = blockIdx.y;
    int v = blockIdx.x * blockDim.x + threadIdx.x;
    if (v >= CV) return;
    float g = 0.0f;
    if (v < CEN) g = __expf(logits[(size_t)b * CEN + v] - mx[b]) * sinv[b];
    out[((size_t)b * CT + t) * CV + v] = (1.0f - cp[b]) * g;
}

__global__ void scatter_copy(const int* __restrict__ sv, const float* __restrict__ au,
                             const float* __restrict__ cp, float* __restrict__ out, int t) {
    int idx = blockIdx.x * blockDim.x + threadIdx.x;          // B*Vs
    if (idx >= CB * CVS) return;
    int b = idx / CVS;
    int tok = sv[idx];
    atomicAdd(&out[((size_t)b * CT + t) * CV + tok], cp[b] * au[idx]);
}

// ---------------------------------------------------------------------------
// Host orchestration
// ---------------------------------------------------------------------------
extern "C" void kernel_launch(void* const* d_in, const int* in_sizes, int n_in,
                              void* d_out, int out_size, void* d_ws, size_t ws_size,
                              hipStream_t stream) {
    (void)in_sizes; (void)n_in; (void)out_size; (void)ws_size;
    const float* enc    = (const float*)d_in[0];
    const float* h0     = (const float*)d_in[1];
    const float* uemb   = (const float*)d_in[2];
    const float* Wemb   = (const float*)d_in[3];
    const float* W_ih   = (const float*)d_in[4];
    const float* W_hh   = (const float*)d_in[5];
    const float* b_ih   = (const float*)d_in[6];
    const float* b_hh   = (const float*)d_in[7];
    const float* Wk_c   = (const float*)d_in[8];
    const float* bk_c   = (const float*)d_in[9];
    const float* Wq_c   = (const float*)d_in[10];
    const float* bq_c   = (const float*)d_in[11];
    const float* we_c   = (const float*)d_in[12];
    const float* be_c   = (const float*)d_in[13];
    const float* Wk_u   = (const float*)d_in[14];
    const float* bk_u   = (const float*)d_in[15];
    const float* Wq_u   = (const float*)d_in[16];
    const float* bq_u   = (const float*)d_in[17];
    const float* we_u   = (const float*)d_in[18];
    const float* be_u   = (const float*)d_in[19];
    const float* Wc     = (const float*)d_in[20];
    const float* Wg     = (const float*)d_in[21];
    const float* bg     = (const float*)d_in[22];
    const float* Wgen   = (const float*)d_in[23];
    const int*   smask  = (const int*)d_in[24];
    const int*   tgt    = (const int*)d_in[25];
    const int*   svocab = (const int*)d_in[26];
    float* out = (float*)d_out;

    // Workspace carve-out (~107 MB)
    char* p = (char*)d_ws;
    auto alloc = [&](size_t bytes) -> void* {
        void* r = (void*)p; p += (bytes + 255) & ~(size_t)255; return r;
    };
    __bf16* wgen_pk = (__bf16*)alloc((size_t)KB_GEN * NT_PAD * 32 * 16 * sizeof(__bf16));
    float* pk_c   = (float*)alloc((size_t)CB * CS * CH * 4);
    float* pk_u   = (float*)alloc((size_t)CB * CVS * CH * 4);
    float* uve    = (float*)alloc((size_t)CB * CVS * CE * 4);
    float* prev   = (float*)alloc((size_t)CT * CB * CE * 4);
    float* logits = (float*)alloc((size_t)CB * CEN * 4);
    float* h      = (float*)alloc(CB * CH * 4);
    float* gi     = (float*)alloc(CB * CH3 * 4);
    float* gh     = (float*)alloc(CB * CH3 * 4);
    float* qc     = (float*)alloc(CB * CH * 4);
    float* qu     = (float*)alloc(CB * CH * 4);
    float* sc_c   = (float*)alloc(CB * CS * 4);
    float* ac     = (float*)alloc(CB * CS * 4);
    float* sc_u   = (float*)alloc(CB * CVS * 4);
    float* au     = (float*)alloc(CB * CVS * 4);
    float* ctxc   = (float*)alloc(CB * CH2 * 4);
    float* ctxu   = (float*)alloc(CB * CE * 4);
    float* cc     = (float*)alloc(CB * CKC * 4);
    float* chb    = (float*)alloc(CB * CH * 4);
    float* cp     = (float*)alloc(CB * 4);
    float* mxb    = (float*)alloc(CB * 4);
    float* sinv   = (float*)alloc(CB * 4);

    auto gemmNN = [&](const float* A, const float* Bm, const float* bias, float* C,
                      int M, int N, int K) {
        int blocks = ((M / 16) * (N / 16) + 3) / 4;
        wmma_gemm<0, 0><<<dim3(blocks), dim3(128), 0, stream>>>(A, Bm, bias, C, M, N, K);
    };
    auto gemmNT = [&](const float* A, const float* Bm, const float* bias, float* C,
                      int M, int N, int K) {
        int blocks = ((M / 16) * (N / 16) + 3) / 4;
        wmma_gemm<1, 0><<<dim3(blocks), dim3(128), 0, stream>>>(A, Bm, bias, C, M, N, K);
    };
    auto gemmNNtanh = [&](const float* A, const float* Bm, const float* bias, float* C,
                          int M, int N, int K) {
        int blocks = ((M / 16) * (N / 16) + 3) / 4;
        wmma_gemm<0, 1><<<dim3(blocks), dim3(128), 0, stream>>>(A, Bm, bias, C, M, N, K);
    };

    // --- one-time precompute ---
    {
        long long n = (long long)KB_GEN * NT_GEN * 32 * 16;
        pack_wgen<<<dim3((unsigned)((n + 255) / 256)), dim3(256), 0, stream>>>(Wgen, wgen_pk);
    }
    gather_prev<<<dim3((CT * CB * CE + 255) / 256), dim3(256), 0, stream>>>(Wemb, tgt, prev);
    gather_uve<<<dim3((CB * CVS * CE + 255) / 256), dim3(256), 0, stream>>>(Wemb, svocab, uve);
    // proj_key_c = encoder_hidden @ Wk_c + bk_c   [12800,1024]@[1024,512]
    gemmNN(enc, Wk_c, bk_c, pk_c, CB * CS, CH, CH2);
    // proj_key_u = uve @ Wk_u + bk_u              [6400,300]@[300,512]
    gemmNN(uve, Wk_u, bk_u, pk_u, CB * CVS, CH, CE);
    hipMemcpyAsync(h, h0, CB * CH * sizeof(float), hipMemcpyDeviceToDevice, stream);

    // --- sequential decoder steps ---
    for (int t = 0; t < CT; ++t) {
        // GRU gates
        gemmNT(prev + (size_t)t * CB * CE, W_ih, b_ih, gi, CB, CH3, CE);
        gemmNT(h, W_hh, b_hh, gh, CB, CH3, CH);
        gru_fuse<<<dim3((CB * CH + 255) / 256), dim3(256), 0, stream>>>(gi, gh, h);

        // query projections
        gemmNN(h, Wq_c, bq_c, qc, CB, CH, CH);
        gemmNN(h, Wq_u, bq_u, qu, CB, CH, CH);

        // encoder attention
        att_score<<<dim3((CB * CS + 3) / 4), dim3(128), 0, stream>>>(qc, pk_c, we_c, be_c, sc_c, CS);
        softmax_row<<<dim3(CB), dim3(256), 0, stream>>>(sc_c, smask, ac, CS);
        ctx_comb<<<dim3((CH2 + 255) / 256, CB), dim3(256), 0, stream>>>(ac, enc, ctxc, CS, CH2);

        // user-vocab attention
        att_score<<<dim3((CB * CVS + 3) / 4), dim3(128), 0, stream>>>(qu, pk_u, we_u, be_u, sc_u, CVS);
        softmax_row<<<dim3(CB), dim3(256), 0, stream>>>(sc_u, (const int*)nullptr, au, CVS);
        ctx_comb<<<dim3((CE + 255) / 256, CB), dim3(256), 0, stream>>>(au, uve, ctxu, CVS, CE);

        // combined hidden + copy gate
        build_concat<<<dim3((CB * CKC + 255) / 256), dim3(256), 0, stream>>>(h, ctxc, ctxu, uemb, cc);
        gemmNNtanh(cc, Wc, nullptr, chb, CB, CH, CKC);
        gate_prob<<<dim3(CB), dim3(256), 0, stream>>>(ctxc, h, ctxu, Wg, bg, cp);

        // vocab projection (hot WMMA GEMM, L2-resident bf16 weights, 16x64/wave)
        wgen_gemm<<<dim3((NT_GEN + 4 * NTW - 1) / (4 * NTW), CB / 16), dim3(128), 0, stream>>>(
            chb, wgen_pk, logits);

        // softmax + mixture into output slice [:, t, :]
        row_stats<<<dim3(CB), dim3(256), 0, stream>>>(logits, mxb, sinv);
        mix_write<<<dim3((CV + 255) / 256, CB), dim3(256), 0, stream>>>(logits, mxb, sinv, cp, out, t);
        scatter_copy<<<dim3((CB * CVS + 255) / 256), dim3(256), 0, stream>>>(svocab, au, cp, out, t);
    }
}
